// SimCLRLoss_63419487093451
// MI455X (gfx1250) — matrix operand order repaired and checked by reading, not compiled, not run on hardware
//
#include <hip/hip_runtime.h>

// SimCLR (NT-Xent) loss, fused GEMM + online softmax for MI455X (gfx1250).
// z = concat(z1,z2): [8192,256] fp32 -> normalize -> f16 -> S = zn*zn^T/0.1
// loss = mean(lse_i - pos_i), lse over the diagonal-masked row.
//
// Since rows are unit vectors, sim <= 1/T = 10 (+f16 eps), so lse uses the
// FIXED shift 10: lse = 10 + log(sum exp(10*c - 10)) -- exact, no running max.
//
// Main kernel: block = 64 rows (4 waves x 16-row A tile in registers), all 4
// waves sweep the same 512 column tiles. Each B tile (16x256 f16 = 8KB) is
// double-buffered in LDS; the *next* tile is staged with CDNA5 async
// load-to-LDS (ASYNCcnt) while the current tile computes: 16 ds_load_b128 as
// one pipelined burst (data-carrying register fence) -> one s_wait_dscnt ->
// 8 back-to-back chained v_wmma_f32_16x16x32_f16.

typedef __attribute__((ext_vector_type(16))) _Float16 v16h;
typedef __attribute__((ext_vector_type(8)))  _Float16 v8h;
typedef __attribute__((ext_vector_type(8)))  float    v8f;
typedef __attribute__((ext_vector_type(4)))  int      v4i;

#define B_ROWS 4096
#define D_DIM  256
#define N_ROWS 8192          // 2*B
#define TILES  (N_ROWS / 16) // 512
#define INV_T  10.0f         // 1 / TEMPERATURE
#define EPS_N  1e-8f

#define ROWS_PER_BLOCK 64
#define LDS_PAD 8
#define LDS_W   (D_DIM + LDS_PAD)   // 264 halfs: 528B row stride -> bank step 4

#if defined(__has_builtin)
#  if __has_builtin(__builtin_amdgcn_global_load_async_to_lds_b128)
#    define HAVE_ASYNC_LDS 1
#  endif
#endif
#ifndef HAVE_ASYNC_LDS
#  define HAVE_ASYNC_LDS 0
#endif

#if HAVE_ASYNC_LDS
// Builtin signature (verified by compiler diagnostic): (global int4*, LDS int4*,
// int offset, int cpol).
typedef __attribute__((address_space(1))) v4i gv4i;   // global int4
typedef __attribute__((address_space(3))) v4i lv4i;   // LDS int4

__device__ __forceinline__ void async_copy_b128(const _Float16* g, _Float16* l) {
    __builtin_amdgcn_global_load_async_to_lds_b128(
        (gv4i*)(v4i*)(void*)const_cast<_Float16*>(g),
        (lv4i*)(v4i*)(void*)l,
        0, 0);
}
__device__ __forceinline__ void async_wait0() {
#if __has_builtin(__builtin_amdgcn_s_wait_asynccnt)
    __builtin_amdgcn_s_wait_asynccnt(0);
#else
    asm volatile("s_wait_asynccnt 0x0" ::: "memory");
#endif
}
#endif

__global__ void simclr_zero_kernel(float* __restrict__ out) { out[0] = 0.0f; }

// One wave (32 lanes) per row: sum of squares -> 1/norm -> write f16 row.
__global__ __launch_bounds__(256) void simclr_normalize_kernel(
    const float* __restrict__ z1, const float* __restrict__ z2,
    _Float16* __restrict__ zn)
{
    int row  = (int)((blockIdx.x * 256u + threadIdx.x) >> 5);
    int lane = threadIdx.x & 31;
    if (row >= N_ROWS) return;

    const float* src = (row < B_ROWS)
        ? (z1 + (size_t)row * D_DIM)
        : (z2 + (size_t)(row - B_ROWS) * D_DIM);

    const float4* p = (const float4*)(src + lane * 8);   // 8 floats per lane
    float4 x0 = p[0];
    float4 x1 = p[1];
    float ss = x0.x*x0.x + x0.y*x0.y + x0.z*x0.z + x0.w*x0.w
             + x1.x*x1.x + x1.y*x1.y + x1.z*x1.z + x1.w*x1.w;
    #pragma unroll
    for (int m = 16; m > 0; m >>= 1) ss += __shfl_xor(ss, m, 32);

    float inv = 1.0f / fmaxf(sqrtf(ss), EPS_N);

    v8h o;
    o[0] = (_Float16)(x0.x * inv); o[1] = (_Float16)(x0.y * inv);
    o[2] = (_Float16)(x0.z * inv); o[3] = (_Float16)(x0.w * inv);
    o[4] = (_Float16)(x1.x * inv); o[5] = (_Float16)(x1.y * inv);
    o[6] = (_Float16)(x1.z * inv); o[7] = (_Float16)(x1.w * inv);
    *(v8h*)(zn + (size_t)row * D_DIM + lane * 8) = o;
}

// pos[i] = dot(zn[i], zn[partner(i)]) / T ; one wave per row.
__global__ __launch_bounds__(256) void simclr_pos_kernel(
    const _Float16* __restrict__ zn, float* __restrict__ pos)
{
    int row  = (int)((blockIdx.x * 256u + threadIdx.x) >> 5);
    int lane = threadIdx.x & 31;
    if (row >= N_ROWS) return;
    int prt = (row < B_ROWS) ? (row + B_ROWS) : (row - B_ROWS);

    v8h a = *(const v8h*)(zn + (size_t)row * D_DIM + lane * 8);
    v8h b = *(const v8h*)(zn + (size_t)prt * D_DIM + lane * 8);
    float d = 0.0f;
    #pragma unroll
    for (int e = 0; e < 8; ++e) d += (float)a[e] * (float)b[e];
    #pragma unroll
    for (int m = 16; m > 0; m >>= 1) d += __shfl_xor(d, m, 32);
    if (lane == 0) pos[row] = d * INV_T;
}

__global__ __launch_bounds__(128, 1) void simclr_main_kernel(
    const _Float16* __restrict__ zn, const float* __restrict__ pos,
    float* __restrict__ out)
{
    __shared__ _Float16 sB[2][16][LDS_W];   // double-buffered B tile

    const int lane = threadIdx.x & 31;
    const int wave = threadIdx.x >> 5;
    const int mrow = lane & 15;     // L % 16
    const int hi   = lane >> 4;     // 0: lanes 0-15, 1: lanes 16-31
    const int i_base = blockIdx.x * ROWS_PER_BLOCK + wave * 16;
    const int tid = (int)threadIdx.x;

    // --- Preload this wave's A tile (16 rows x 256 K) into registers.
    v16h A[8];
    {
        const _Float16* arow = zn + (size_t)(i_base + mrow) * D_DIM;
        const int abase = hi * 8;
        #pragma unroll
        for (int kb = 0; kb < 8; ++kb) {
            v8h c0 = *(const v8h*)(arow + kb * 32 + abase);
            v8h c1 = *(const v8h*)(arow + kb * 32 + abase + 16);
            #pragma unroll
            for (int e = 0; e < 8; ++e) { A[kb][e] = c0[e]; A[kb][8 + e] = c1[e]; }
        }
    }

    // Per-lane shifted sum-exp state: 8 rows (r -> row r + 8*hi).
    float rsum[8];
    #pragma unroll
    for (int r = 0; r < 8; ++r) rsum[r] = 0.0f;

    // --- Stage tile 0 into buffer 0 (chunk c = tid + 128*q : 512 x 16B).
    #pragma unroll
    for (int q = 0; q < 4; ++q) {
        int c = tid + 128 * q;
        int r = c >> 5, col = (c & 31) * 8;
#if HAVE_ASYNC_LDS
        async_copy_b128(zn + (size_t)r * D_DIM + col, &sB[0][r][col]);
#else
        float4 v = *(const float4*)(zn + (size_t)r * D_DIM + col);
        *(float4*)&sB[0][r][col] = v;
#endif
    }
#if HAVE_ASYNC_LDS
    async_wait0();
#endif
    __syncthreads();

    const int dtile = i_base >> 4;  // the one tile containing this wave's diagonal
    const int boff = hi * 8;        // B 32x16: lanes 0-15 K=0..7,16..23, 16-31 K=8..15,24..31
    for (int jt = 0; jt < TILES; ++jt) {
        const int p = jt & 1;
        const bool more = (jt + 1 < TILES);

#if !HAVE_ASYNC_LDS
        float4 pre0, pre1, pre2, pre3;
#endif
        // 1) Stage tile jt+1 into the other buffer (overlaps with compute).
        if (more) {
            const int jb = (jt + 1) * 16;
#if HAVE_ASYNC_LDS
            #pragma unroll
            for (int q = 0; q < 4; ++q) {
                int c = tid + 128 * q;
                int r = c >> 5, col = (c & 31) * 8;
                async_copy_b128(zn + (size_t)(jb + r) * D_DIM + col,
                                &sB[p ^ 1][r][col]);
            }
#else
            const _Float16* gsrc = zn + (size_t)jb * D_DIM;
            pre0 = *(const float4*)(gsrc + (size_t)((tid      ) >> 5) * D_DIM + ((tid      ) & 31) * 8);
            pre1 = *(const float4*)(gsrc + (size_t)((tid + 128) >> 5) * D_DIM + ((tid + 128) & 31) * 8);
            pre2 = *(const float4*)(gsrc + (size_t)((tid + 256) >> 5) * D_DIM + ((tid + 256) & 31) * 8);
            pre3 = *(const float4*)(gsrc + (size_t)((tid + 384) >> 5) * D_DIM + ((tid + 384) & 31) * 8);
#endif
        }

        // 2) Pull the whole B tile from LDS as one 16-load burst ...
        const _Float16* brow = &sB[p][mrow][0];
        v16h Bv[8];
        #pragma unroll
        for (int kb = 0; kb < 8; ++kb) {
            v8h b0 = *(const v8h*)(brow + kb * 32 + boff);
            v8h b1 = *(const v8h*)(brow + kb * 32 + boff + 16);
            #pragma unroll
            for (int e = 0; e < 8; ++e) { Bv[kb][e] = b0[e]; Bv[kb][8 + e] = b1[e]; }
        }
        // Data-carrying fence: all 8 fragments become asm outputs, so the 16
        // ds_loads must complete before this point (one s_wait_dscnt, loads
        // pipelined), they occupy distinct VGPR octets (no recycling), and
        // the WMMA chain below cannot be hoisted across.
        asm volatile(""
            : "+v"(Bv[0]), "+v"(Bv[1]), "+v"(Bv[2]), "+v"(Bv[3]),
              "+v"(Bv[4]), "+v"(Bv[5]), "+v"(Bv[6]), "+v"(Bv[7]));

        // ... then 8 back-to-back WMMAs chained on the f32 accumulator.
        v8f cacc = {};
        #pragma unroll
        for (int kb = 0; kb < 8; ++kb) {
            cacc = __builtin_amdgcn_wmma_f32_16x16x32_f16(
                       false, A[kb], false, Bv[kb], (short)0, cacc, false, false);
        }

        // 3) Fold: rsum += exp(10*c - 10). Fixed shift is exact (cos <= 1).
        //    Diagonal lives in exactly one wave-uniform tile -> branch it out.
        if (jt == dtile) {
            #pragma unroll
            for (int r = 0; r < 8; ++r) {
                const bool diag = ((r + hi * 8) == mrow);
                float e = __expf(fmaf(cacc[r], INV_T, -INV_T));
                rsum[r] += diag ? 0.0f : e;
            }
        } else {
            #pragma unroll
            for (int r = 0; r < 8; ++r)
                rsum[r] += __expf(fmaf(cacc[r], INV_T, -INV_T));
        }

        // 4) Make the staged tile visible, then barrier.
        if (more) {
#if HAVE_ASYNC_LDS
            async_wait0();
#else
            *(float4*)&sB[p ^ 1][(tid      ) >> 5][((tid      ) & 31) * 8] = pre0;
            *(float4*)&sB[p ^ 1][(tid + 128) >> 5][((tid + 128) & 31) * 8] = pre1;
            *(float4*)&sB[p ^ 1][(tid + 256) >> 5][((tid + 256) & 31) * 8] = pre2;
            *(float4*)&sB[p ^ 1][(tid + 384) >> 5][((tid + 384) & 31) * 8] = pre3;
#endif
        }
        __syncthreads();
    }

    // Sum across the 16 lanes holding the same rows (xor stays in each half).
    #pragma unroll
    for (int m = 1; m < 16; m <<= 1) {
        #pragma unroll
        for (int r = 0; r < 8; ++r) rsum[r] += __shfl_xor(rsum[r], m, 32);
    }

    // lse = 10 + log(sum); subtract positive logit; one atomic per wave.
    float part = 0.0f;
    #pragma unroll
    for (int r = 0; r < 8; ++r)
        part += (INV_T + __logf(rsum[r])) - pos[i_base + r + hi * 8];
    part += __shfl_xor(part, 16, 32);
    if (lane == 0) atomicAdd(out, part * (1.0f / (float)N_ROWS));
}

extern "C" void kernel_launch(void* const* d_in, const int* in_sizes, int n_in,
                              void* d_out, int out_size, void* d_ws, size_t ws_size,
                              hipStream_t stream) {
    (void)in_sizes; (void)n_in; (void)out_size; (void)ws_size;
    const float* z1 = (const float*)d_in[0];
    const float* z2 = (const float*)d_in[1];
    float* out = (float*)d_out;

    _Float16* zn = (_Float16*)d_ws;                                  // 4 MB
    float* pos = (float*)((char*)d_ws + (size_t)N_ROWS * D_DIM * sizeof(_Float16));

    simclr_zero_kernel<<<1, 1, 0, stream>>>(out);
    simclr_normalize_kernel<<<N_ROWS / 8, 256, 0, stream>>>(z1, z2, zn);
    simclr_pos_kernel<<<N_ROWS / 8, 256, 0, stream>>>(zn, pos);
    simclr_main_kernel<<<N_ROWS / ROWS_PER_BLOCK, 128, 0, stream>>>(zn, pos, out);
}
